// CG_MSA_M_65171833749563
// MI455X (gfx1250) — matrix operand order repaired and checked by reading, compiled, not run on hardware
//
#include <hip/hip_runtime.h>
#include <hip/hip_bf16.h>

#define BATCH 8
#define CDIM  96
#define NHEAD 6
#define HH    128
#define WW    128
#define NSP   (HH * WW)     // 16384 spatial positions
#define PH    (HH + 2)      // padded height for fuse conv
#define PW    144           // padded row stride (>= WW+2, 32B-aligned rows)

typedef __attribute__((ext_vector_type(16))) __bf16 v16bf;
typedef __attribute__((ext_vector_type(8)))  float  v8f;

__device__ __forceinline__ __bf16 f2bf(float f) { return (__bf16)f; }
__device__ __forceinline__ float  bf2f(__bf16 b) { return (float)b; }

// A-matrix K swizzle for 16-bit 16x32 A fragments (ISA 7.12.2):
// element e -> VGPR e/2 slot e%2: v<4 -> K = 2v+bit+8*half ; else 16+2(v-4)+bit+8*half
__device__ __forceinline__ int kswz(int e, int half) {
  int v = e >> 1, bit = e & 1;
  return (v < 4) ? (2 * v + bit + 8 * half) : (16 + 2 * (v - 4) + bit + 8 * half);
}

__device__ __forceinline__ v16bf ld16bf_u(const __bf16* p) {  // unaligned 32B load
  v16bf r;
  __builtin_memcpy(&r, p, sizeof(r));
  return r;
}

// =====================================================================
// fp32 -> bf16 tensor conversion (one pass; feeds all WMMA B operands)
// =====================================================================
__global__ void __launch_bounds__(256)
k_cvt_bf16(const float* __restrict__ X, __bf16* __restrict__ Y, size_t n)
{
  size_t i = ((size_t)blockIdx.x * 256 + threadIdx.x) * 4;
  if (i >= n) return;
  float4 f = *(const float4*)(X + i);
  Y[i + 0] = f2bf(f.x);
  Y[i + 1] = f2bf(f.y);
  Y[i + 2] = f2bf(f.z);
  Y[i + 3] = f2bf(f.w);
}

// =====================================================================
// Pre-swizzle weights into WMMA A-fragment order:
//   Af[((mt*nkc + kc)*32 + lane)*16 + e]  (bf16)
// mode 0: W row-major [M][K]
// mode 1: w_fuse OIHW with K = tap*192 + ic  -> src (m*192+ic)*9 + tap
// =====================================================================
__global__ void __launch_bounds__(256)
k_prep_afrag(const float* __restrict__ Wsrc, __bf16* __restrict__ Af,
             int M, int K, int mode)
{
  int idx = blockIdx.x * 256 + threadIdx.x;
  if (idx >= M * K) return;
  int nkc  = K >> 5;
  int e    = idx & 15;
  int lane = (idx >> 4) & 31;
  int rest = idx >> 9;
  int kc   = rest % nkc;
  int mt   = rest / nkc;
  int half = lane >> 4, l16 = lane & 15;
  int m = mt * 16 + l16;
  int k = kc * 32 + kswz(e, half);
  float v;
  if (mode == 0) v = Wsrc[(size_t)m * K + k];
  else { int tap = k / 192, ic = k % 192; v = Wsrc[((size_t)m * 192 + ic) * 9 + tap]; }
  Af[idx] = f2bf(v);
}

// =====================================================================
// Zero a bf16 buffer (16B vector stores) — border init for padded vcat
// =====================================================================
__global__ void __launch_bounds__(256)
k_zero_bf16(__bf16* __restrict__ P, size_t n)
{
  size_t i = ((size_t)blockIdx.x * 256 + threadIdx.x) * 8;
  if (i >= n) return;
  uint4 z = {0u, 0u, 0u, 0u};
  *(uint4*)(P + i) = z;
}

// =====================================================================
// 1x1 conv as GEMM: Out[b,m,n] = sum_k W[m,k]*X[b,k,n]; bf16 in/out,
// f32 accumulate. K rows < CDIM from X0, rows >= CDIM from X1.
// Wave computes a 16(M) x 64(N) tile.
// =====================================================================
__global__ void __launch_bounds__(256)
k_gemm1x1(const __bf16* __restrict__ X0, const __bf16* __restrict__ X1,
          const __bf16* __restrict__ Af, __bf16* __restrict__ Out,
          int M, int K)
{
  const int lane = threadIdx.x & 31;
  const int wv   = threadIdx.x >> 5;
  const int half = lane >> 4, l16 = lane & 15;
  const int b  = blockIdx.z;
  const int n0 = blockIdx.x * 64;
  const int m0 = blockIdx.y * 128 + wv * 16;
  if (m0 >= M) return;

  const int nkc = K >> 5;
  const int mt  = m0 >> 4;
  v8f acc[4] = {};
  for (int kc = 0; kc < nkc; ++kc) {
    // A fragment: one aligned 32B vector load (pre-swizzled)
    v16bf a = *(const v16bf*)(Af + (((size_t)mt * nkc + kc) * 32 + lane) * 16);

    // B fragment: element e = K row kc*32+kswz(e,half), lane%16 = spatial column
    int krow = kc * 32 + lane;
    const __bf16* xb = ((krow < CDIM)
        ? (X0 + ((size_t)b * CDIM + krow) * NSP)
        : (X1 + ((size_t)b * CDIM + (krow - CDIM)) * NSP)) + n0;

    if (kc + 1 < nkc) {
      int kr2 = krow + 32;
      const __bf16* pr = ((kr2 < CDIM)
          ? (X0 + ((size_t)b * CDIM + kr2) * NSP)
          : (X1 + ((size_t)b * CDIM + (kr2 - CDIM)) * NSP)) + n0;
      __builtin_prefetch(pr, 0, 1);   // -> global_prefetch_b8
    }

#pragma unroll
    for (int s = 0; s < 4; ++s) {
      v16bf bb;
#pragma unroll
      for (int e = 0; e < 16; ++e) {
        int kr = kc * 32 + kswz(e, half);
        const __bf16* src = (kr < CDIM)
            ? (X0 + ((size_t)b * CDIM + kr) * NSP)
            : (X1 + ((size_t)b * CDIM + (kr - CDIM)) * NSP);
        bb[e] = src[n0 + s * 16 + l16];
      }
      acc[s] = __builtin_amdgcn_wmma_f32_16x16x32_bf16(
          false, a, false, bb, (short)0, acc[s], false, false);
    }
  }
#pragma unroll
  for (int s = 0; s < 4; ++s)
#pragma unroll
    for (int r = 0; r < 8; ++r)
      Out[((size_t)b * M + m0 + r + 8 * half) * NSP + n0 + s * 16 + l16] =
          f2bf(acc[s][r]);
}

// =====================================================================
// Depthwise 3x3, pad=1 (192 ch). First 96 out ch -> OutA (q or k);
// last 96 -> padded Vcat at channel offset voff (interior only).
// =====================================================================
__global__ void __launch_bounds__(256)
k_dw3x3(const __bf16* __restrict__ T, const float* __restrict__ Wd,
        __bf16* __restrict__ OutA, __bf16* __restrict__ VcatP, int voff)
{
  size_t idx = (size_t)blockIdx.x * 256 + threadIdx.x;
  const size_t total = (size_t)BATCH * 2 * CDIM * NSP;
  if (idx >= total) return;
  int x = (int)(idx % WW);
  int y = (int)((idx / WW) % HH);
  int c = (int)((idx / NSP) % (2 * CDIM));
  int b = (int)(idx / ((size_t)2 * CDIM * NSP));
  const __bf16* src = T + ((size_t)b * 2 * CDIM + c) * NSP;
  float acc = 0.f;
#pragma unroll
  for (int ky = 0; ky < 3; ++ky) {
    int yy = y + ky - 1;
    if (yy < 0 || yy >= HH) continue;
#pragma unroll
    for (int kx = 0; kx < 3; ++kx) {
      int xx = x + kx - 1;
      if (xx < 0 || xx >= WW) continue;
      acc += bf2f(src[yy * WW + xx]) * Wd[(c * 3 + ky) * 3 + kx];
    }
  }
  __bf16 r = f2bf(acc);
  if (c < CDIM) OutA[((size_t)b * CDIM + c) * NSP + y * WW + x] = r;
  else {
    int ch = voff + (c - CDIM);
    VcatP[(((size_t)b * 2 * CDIM + ch) * PH + y + 1) * PW + x + 1] = r;
  }
}

// =====================================================================
// v_fuse 3x3 (M=96, K=1728) implicit WMMA GEMM on the padded vcat.
// K = tap*192 + ic -> each 32-chunk has a uniform tap; no bounds checks.
// =====================================================================
__global__ void __launch_bounds__(256)
k_fuse3x3(const __bf16* __restrict__ VcP, const __bf16* __restrict__ Af,
          const float* __restrict__ Bf, __bf16* __restrict__ Vout)
{
  const int lane = threadIdx.x & 31, wv = threadIdx.x >> 5;
  const int half = lane >> 4, l16 = lane & 15;
  if (wv >= CDIM / 16) return;   // 6 active waves
  const int b  = blockIdx.z;
  const int y  = blockIdx.y;
  const int x0 = blockIdx.x * 64;
  const int nkc = 54;            // 1728 / 32

  v8f acc[4] = {};
  for (int kc = 0; kc < nkc; ++kc) {
    int tap = kc / 6;
    int icb = (kc % 6) * 32;
    int dy = tap / 3 - 1, dx = tap % 3 - 1;

    v16bf a = *(const v16bf*)(Af + (((size_t)wv * nkc + kc) * 32 + lane) * 16);

    int ic = icb + lane;
    const __bf16* row = VcP + (((size_t)b * 2 * CDIM + ic) * PH + y + dy + 1) * PW
                        + (x0 + dx + 1);
#pragma unroll
    for (int s = 0; s < 4; ++s) {
      v16bf bb;
#pragma unroll
      for (int e = 0; e < 16; ++e)
        bb[e] = VcP[(((size_t)b * 2 * CDIM + icb + kswz(e, half)) * PH + y + dy + 1) * PW
                    + (x0 + dx + 1) + s * 16 + l16];
      acc[s] = __builtin_amdgcn_wmma_f32_16x16x32_bf16(
          false, a, false, bb, (short)0, acc[s], false, false);
    }
  }
#pragma unroll
  for (int s = 0; s < 4; ++s)
#pragma unroll
    for (int r = 0; r < 8; ++r) {
      int oc = wv * 16 + r + 8 * half;
      Vout[((size_t)b * CDIM + oc) * NSP + y * WW + x0 + s * 16 + l16] =
          f2bf(acc[s][r] + Bf[oc]);
    }
}

// =====================================================================
// Per-channel L2 norm over spatial positions
// =====================================================================
__global__ void __launch_bounds__(256)
k_rownorm(const __bf16* __restrict__ T, float* __restrict__ Out)
{
  const __bf16* row = T + (size_t)blockIdx.x * NSP;
  float s = 0.f;
  for (int n = threadIdx.x; n < NSP; n += 256) { float v = bf2f(row[n]); s += v * v; }
  __shared__ float red[256];
  red[threadIdx.x] = s; __syncthreads();
  for (int off = 128; off > 0; off >>= 1) {
    if (threadIdx.x < off) red[threadIdx.x] += red[threadIdx.x + off];
    __syncthreads();
  }
  if (threadIdx.x == 0) Out[blockIdx.x] = sqrtf(red[0]);
}

// =====================================================================
// Raw channel-channel logits S[b,h,c,d] = sum_n q[c,n]*k[d,n]
// =====================================================================
__global__ void __launch_bounds__(256)
k_qk(const __bf16* __restrict__ Q, const __bf16* __restrict__ Kt,
     float* __restrict__ S)
{
  const int c = blockIdx.x, h = blockIdx.y, b = blockIdx.z;
  const __bf16* qrow  = Q  + ((size_t)b * CDIM + h * 16 + c) * NSP;
  const __bf16* kbase = Kt + ((size_t)b * CDIM + h * 16) * NSP;
  float acc[16] = {};
  for (int n = threadIdx.x; n < NSP; n += 256) {
    float qv = bf2f(qrow[n]);
#pragma unroll
    for (int d = 0; d < 16; ++d) acc[d] += qv * bf2f(kbase[(size_t)d * NSP + n]);
  }
  __shared__ float red[256];
  float* srow = S + ((size_t)(b * NHEAD + h) * 16 + c) * 16;
#pragma unroll 1
  for (int d = 0; d < 16; ++d) {
    red[threadIdx.x] = acc[d]; __syncthreads();
    for (int off = 128; off > 0; off >>= 1) {
      if (threadIdx.x < off) red[threadIdx.x] += red[threadIdx.x + off];
      __syncthreads();
    }
    if (threadIdx.x == 0) srow[d] = red[0];
    __syncthreads();
  }
}

// =====================================================================
// attn = softmax_d( S / (max(|q|,eps)*max(|k|,eps)) * temperature[h] )
// =====================================================================
__global__ void __launch_bounds__(256)
k_softmax(const float* __restrict__ S, const float* __restrict__ Nq,
          const float* __restrict__ Nk, const float* __restrict__ Temp,
          float* __restrict__ A)
{
  const int h = blockIdx.x, b = blockIdx.y;
  const int tid = threadIdx.x, c = tid >> 4, d = tid & 15;
  size_t base = (size_t)(b * NHEAD + h) * 256;
  float nq = fmaxf(Nq[b * CDIM + h * 16 + c], 1e-12f);
  float nk = fmaxf(Nk[b * CDIM + h * 16 + d], 1e-12f);
  float l = S[base + tid] / (nq * nk) * Temp[h];
  __shared__ float sm[256];
  sm[tid] = l; __syncthreads();
  for (int off = 8; off > 0; off >>= 1) {
    if (d < off) sm[tid] = fmaxf(sm[tid], sm[tid + off]);
    __syncthreads();
  }
  float mx = sm[c * 16]; __syncthreads();
  float e = expf(l - mx);
  sm[tid] = e; __syncthreads();
  for (int off = 8; off > 0; off >>= 1) {
    if (d < off) sm[tid] += sm[tid + off];
    __syncthreads();
  }
  A[base + tid] = e / sm[c * 16];
}

// =====================================================================
// out[b,h,c,n] = sum_d attn[b,h,c,d] * v[b,h,d,n]
// =====================================================================
__global__ void __launch_bounds__(256)
k_attnv(const float* __restrict__ A, const __bf16* __restrict__ V,
        __bf16* __restrict__ O)
{
  const int b = blockIdx.y;
  const int n = blockIdx.x * 256 + threadIdx.x;
  __shared__ float at[NHEAD * 256];
  for (int i = threadIdx.x; i < NHEAD * 256; i += 256)
    at[i] = A[(size_t)b * NHEAD * 256 + i];
  __syncthreads();
#pragma unroll 1
  for (int h = 0; h < NHEAD; ++h) {
    float vv[16];
#pragma unroll
    for (int d = 0; d < 16; ++d)
      vv[d] = bf2f(V[((size_t)b * CDIM + h * 16 + d) * NSP + n]);
#pragma unroll
    for (int c2 = 0; c2 < 16; ++c2) {
      float o = 0.f;
#pragma unroll
      for (int d = 0; d < 16; ++d) o += at[h * 256 + c2 * 16 + d] * vv[d];
      O[((size_t)b * CDIM + h * 16 + c2) * NSP + n] = f2bf(o);
    }
  }
}

// =====================================================================
// project_out 1x1 (M=192, K=96) + xy_pos residual -> f32 output
// =====================================================================
__global__ void __launch_bounds__(256)
k_proj(const __bf16* __restrict__ Xa, const __bf16* __restrict__ Af,
       const __bf16* __restrict__ Pos, float* __restrict__ Out)
{
  const int lane = threadIdx.x & 31, wv = threadIdx.x >> 5;
  const int half = lane >> 4, l16 = lane & 15;
  const int b  = blockIdx.z;
  const int n0 = blockIdx.x * 64;
  const int M  = 2 * CDIM;
  const int m0 = blockIdx.y * 128 + wv * 16;
  if (m0 >= M) return;
  const int nkc = 3, mt = m0 >> 4;

  v8f acc[4] = {};
#pragma unroll
  for (int kc = 0; kc < nkc; ++kc) {
    v16bf a = *(const v16bf*)(Af + (((size_t)mt * nkc + kc) * 32 + lane) * 16);
    const __bf16* xr = Xa + ((size_t)b * CDIM + kc * 32 + lane) * NSP + n0;
#pragma unroll
    for (int s = 0; s < 4; ++s) {
      v16bf bb;
#pragma unroll
      for (int e = 0; e < 16; ++e)
        bb[e] = Xa[((size_t)b * CDIM + kc * 32 + kswz(e, half)) * NSP + n0 + s * 16 + l16];
      acc[s] = __builtin_amdgcn_wmma_f32_16x16x32_bf16(
          false, a, false, bb, (short)0, acc[s], false, false);
    }
  }
#pragma unroll
  for (int s = 0; s < 4; ++s)
#pragma unroll
    for (int r = 0; r < 8; ++r) {
      size_t o = ((size_t)b * M + m0 + r + 8 * half) * NSP + n0 + s * 16 + l16;
      Out[o] = acc[s][r] + bf2f(Pos[o]);
    }
}

// =====================================================================
extern "C" void kernel_launch(void* const* d_in, const int* in_sizes, int n_in,
                              void* d_out, int out_size, void* d_ws, size_t ws_size,
                              hipStream_t stream) {
  (void)in_sizes; (void)n_in; (void)out_size; (void)ws_size;
  const float* x       = (const float*)d_in[0];
  const float* y       = (const float*)d_in[1];
  const float* w_pos   = (const float*)d_in[2];
  const float* w_qv    = (const float*)d_in[3];
  const float* w_qv_dw = (const float*)d_in[4];
  const float* w_kv    = (const float*)d_in[5];
  const float* w_kv_dw = (const float*)d_in[6];
  const float* w_proj  = (const float*)d_in[7];
  const float* w_fuse  = (const float*)d_in[8];
  const float* b_fuse  = (const float*)d_in[9];
  const float* temp    = (const float*)d_in[10];
  float* out = (float*)d_out;

  const size_t nBCN  = (size_t)BATCH * CDIM * NSP;           // 96-ch tensor elems
  const size_t nPAD  = (size_t)BATCH * 2 * CDIM * PH * PW;   // padded vcat elems
  __bf16* ws    = (__bf16*)d_ws;
  __bf16* xb    = ws;                  //   nBCN : x in bf16
  __bf16* yb    = xb   + nBCN;         //   nBCN : y in bf16
  __bf16* pos   = yb   + nBCN;         // 2*nBCN : xy_pos
  __bf16* t_qv  = pos  + 2 * nBCN;     // 2*nBCN : conv1x1(x, w_qv)
  __bf16* t_kv  = t_qv + 2 * nBCN;     // 2*nBCN : conv1x1(y, w_kv)
  __bf16* qb    = t_kv + 2 * nBCN;     //   nBCN : q
  __bf16* kb    = qb   + nBCN;         //   nBCN : k
  __bf16* vb    = kb   + nBCN;         //   nBCN : fused v
  __bf16* aout  = vb   + nBCN;         //   nBCN : attention out
  __bf16* vcatP = aout + nBCN;         //   nPAD : padded concat(v0, v_)
  float*  S     = (float*)(vcatP + nPAD);
  float*  nq    = S  + (size_t)BATCH * NHEAD * 256;
  float*  nk    = nq + BATCH * CDIM;
  float*  attn  = nk + BATCH * CDIM;
  __bf16* afPos  = (__bf16*)(attn + (size_t)BATCH * NHEAD * 256);
  __bf16* afQv   = afPos  + 192 * 192;
  __bf16* afKv   = afQv   + 192 * 96;
  __bf16* afProj = afKv   + 192 * 96;
  __bf16* afFuse = afProj + 192 * 96;   // 96*1728 elems

  dim3 blk(256);

  // --- prep: bf16 inputs + pre-swizzled A fragments ---
  k_cvt_bf16<<<dim3((unsigned)((nBCN / 4 + 255) / 256)), blk, 0, stream>>>(x, xb, nBCN);
  k_cvt_bf16<<<dim3((unsigned)((nBCN / 4 + 255) / 256)), blk, 0, stream>>>(y, yb, nBCN);
  k_prep_afrag<<<dim3((192 * 192 + 255) / 256), blk, 0, stream>>>(w_pos,  afPos,  192, 192, 0);
  k_prep_afrag<<<dim3((192 * 96 + 255) / 256),  blk, 0, stream>>>(w_qv,   afQv,   192, 96, 0);
  k_prep_afrag<<<dim3((192 * 96 + 255) / 256),  blk, 0, stream>>>(w_kv,   afKv,   192, 96, 0);
  k_prep_afrag<<<dim3((192 * 96 + 255) / 256),  blk, 0, stream>>>(w_proj, afProj, 192, 96, 0);
  k_prep_afrag<<<dim3((96 * 1728 + 255) / 256), blk, 0, stream>>>(w_fuse, afFuse, 96, 1728, 1);
  k_zero_bf16<<<dim3((unsigned)((nPAD / 8 + 255) / 256)), blk, 0, stream>>>(vcatP, nPAD);

  // --- 1x1 convs (WMMA GEMMs) ---
  dim3 gGemm(NSP / 64, 2, BATCH);
  k_gemm1x1<<<gGemm, blk, 0, stream>>>(xb, yb, afPos, pos,  2 * CDIM, 2 * CDIM);
  k_gemm1x1<<<gGemm, blk, 0, stream>>>(xb, xb, afQv,  t_qv, 2 * CDIM, CDIM);
  k_gemm1x1<<<gGemm, blk, 0, stream>>>(yb, yb, afKv,  t_kv, 2 * CDIM, CDIM);

  // --- depthwise 3x3 -> q, k, padded concat(v0, v_) ---
  const size_t totalDW = (size_t)BATCH * 2 * CDIM * NSP;
  dim3 gDW((unsigned)((totalDW + 255) / 256));
  k_dw3x3<<<gDW, blk, 0, stream>>>(t_qv, w_qv_dw, qb, vcatP, CDIM); // v_ -> ch 96..191
  k_dw3x3<<<gDW, blk, 0, stream>>>(t_kv, w_kv_dw, kb, vcatP, 0);    // v0 -> ch 0..95

  // --- v_fuse 3x3 (WMMA implicit GEMM) ---
  k_fuse3x3<<<dim3(WW / 64, HH, BATCH), blk, 0, stream>>>(vcatP, afFuse, b_fuse, vb);

  // --- attention (tiny): norms, logits, softmax, attn@v ---
  k_rownorm<<<dim3(BATCH * CDIM), blk, 0, stream>>>(qb, nq);
  k_rownorm<<<dim3(BATCH * CDIM), blk, 0, stream>>>(kb, nk);
  k_qk<<<dim3(16, NHEAD, BATCH), blk, 0, stream>>>(qb, kb, S);
  k_softmax<<<dim3(NHEAD, BATCH), blk, 0, stream>>>(S, nq, nk, temp, attn);
  k_attnv<<<dim3(NSP / 256, BATCH), blk, 0, stream>>>(attn, vb, aout);

  // --- project_out + xy_pos (WMMA GEMM, overwrites d_out) ---
  k_proj<<<gGemm, blk, 0, stream>>>(aout, afProj, pos, out);
}
